// GradMimicryNetwork_867583393951
// MI455X (gfx1250) — compile-verified, matrix operand order checked by reading, not run on hardware
//
#include <hip/hip_runtime.h>
#include <hip/hip_bf16.h>

// ---------------------------------------------------------------------------
// LIF SNN restructured:
//   1) transpose/convert spikes (B,784,T)f32 -> X[(t*B+b)][k]f16, K padded 800
//      (async global->LDS path on interior tiles, ASYNCcnt-tracked)
//   2) GEMM1 (WMMA f16, double-buffered): I_h = X @ relu(W_ih)
//   3) hidden recurrence + sigmoid -> s_h (f16)
//   4) GEMM2 (WMMA f16): I_o = s_h @ relu(W_ho) padN16
//   5) output recurrence -> spikes + firing rates
// ---------------------------------------------------------------------------

typedef _Float16 v16h __attribute__((ext_vector_type(16)));
typedef _Float16 v8h  __attribute__((ext_vector_type(8)));
typedef float    v8f  __attribute__((ext_vector_type(8)));

#define BATCH 128
#define NI    784
#define NH    256
#define NOUT  10
#define TT    500
#define KP    800                 // padded K (input->hidden), 25 tiles of 32
#define MROWS (BATCH * TT)        // 64000 GEMM rows

#if defined(__has_builtin)
#if __has_builtin(__builtin_amdgcn_global_load_async_to_lds_b32)
#define HAVE_ASYNC_G2L 1
#endif
#endif
#ifndef HAVE_ASYNC_G2L
#define HAVE_ASYNC_G2L 0
#endif

union V16 { v16h v; v8h h[2]; };

__device__ __forceinline__ float sigm5(float v) {
    // sigmoid(5*(v - 1))
    return 1.0f / (1.0f + __expf(-5.0f * (v - 1.0f)));
}

#if HAVE_ASYNC_G2L
typedef __attribute__((address_space(1))) int g_int;
typedef __attribute__((address_space(3))) int l_int;
__device__ __forceinline__ void async_g2l_b32(const void* g, void* l) {
    __builtin_amdgcn_global_load_async_to_lds_b32(
        (g_int*)(g), (l_int*)(l), 0, 0);
}
__device__ __forceinline__ void wait_async0() {
#if defined(__has_builtin) && __has_builtin(__builtin_amdgcn_s_wait_asynccnt)
    __builtin_amdgcn_s_wait_asynccnt(0);
#else
    asm volatile("s_wait_asynccnt 0x0" ::: "memory");
#endif
}
#endif

// ---- weight prep: Wt[n][k] = relu(w1[k][n]) as f16, k padded to KP ---------
__global__ void k_prep_w1(const float* __restrict__ w1, _Float16* __restrict__ Wt) {
    int idx = blockIdx.x * blockDim.x + threadIdx.x;   // 256*800 threads
    int n = idx / KP;
    int k = idx - n * KP;
    float v = (k < NI) ? fmaxf(w1[(size_t)k * NH + n], 0.0f) : 0.0f;
    Wt[idx] = (_Float16)v;
}

// ---- Wot[n][k] = relu(w2[k][n]) as f16, n padded 10 -> 16 ------------------
__global__ void k_prep_w2(const float* __restrict__ w2, _Float16* __restrict__ Wot) {
    int idx = blockIdx.x * blockDim.x + threadIdx.x;   // 16*256 threads
    int n = idx / NH;
    int k = idx - n * NH;
    float v = (n < NOUT) ? fmaxf(w2[(size_t)k * NOUT + n], 0.0f) : 0.0f;
    Wot[idx] = (_Float16)v;
}

// ---- LDS-tiled transpose: (b,i,t)f32 -> X[(t*B+b)][i]f16, pad i to 800 -----
__global__ void k_transpose(const float* __restrict__ in, _Float16* __restrict__ Xh) {
    __shared__ float tile[32][33];                     // +1 pad: no bank conflicts
    const int i0 = blockIdx.x * 32;                    // feature tile (25)
    const int t0 = blockIdx.y * 32;                    // time tile    (16)
    const int b  = blockIdx.z;                         // batch        (128)
    const int tx = threadIdx.x;                        // 0..31 -> t on load
    const int ty = threadIdx.y;                        // 0..7
    const float* src = in + (size_t)b * (NI * TT);

#if HAVE_ASYNC_G2L
    // Interior tiles (no bounds handling needed): async global->LDS copy,
    // tracked by ASYNCcnt (gfx1250 async path). Uniform branch per block.
    if (blockIdx.x < 24 && blockIdx.y < 15) {
#pragma unroll
        for (int q = 0; q < 4; ++q) {
            int i = i0 + ty + 8 * q;
            int t = t0 + tx;
            async_g2l_b32(src + (size_t)i * TT + t, &tile[ty + 8 * q][tx]);
        }
        wait_async0();
        __syncthreads();
    } else
#endif
    {
#pragma unroll
        for (int q = 0; q < 4; ++q) {
            int i = i0 + ty + 8 * q;
            int t = t0 + tx;
            float v = 0.0f;
            if (i < NI && t < TT) v = src[(size_t)i * TT + t];   // coalesced over t
            tile[ty + 8 * q][tx] = v;
        }
        __syncthreads();
    }

#pragma unroll
    for (int q = 0; q < 4; ++q) {
        int t = t0 + ty + 8 * q;
        int i = i0 + tx;
        if (t < TT) {                                  // coalesced over i
            size_t m = (size_t)t * BATCH + b;
            Xh[m * KP + i] = (_Float16)tile[tx][ty + 8 * q];
        }
    }
}

// ---- GEMM1: I_h(64000x256) = X(64000x800) @ Wt^T ; wave = 16x64 strip ------
// Double-buffered over k-tiles so loads overlap the WMMA pipe.
#define G1_LOAD(A_, B0_, B1_, B2_, B3_, KO_)                                   \
    do {                                                                       \
        (A_).h[0]  = *(const v8h*)(pA + (KO_));                                \
        (A_).h[1]  = *(const v8h*)(pA + (KO_) + 16);                           \
        (B0_).h[0] = *(const v8h*)(pB + (KO_));                                \
        (B0_).h[1] = *(const v8h*)(pB + (KO_) + 16);                           \
        (B1_).h[0] = *(const v8h*)(pB + 16 * KP + (KO_));                      \
        (B1_).h[1] = *(const v8h*)(pB + 16 * KP + (KO_) + 16);                 \
        (B2_).h[0] = *(const v8h*)(pB + 32 * KP + (KO_));                      \
        (B2_).h[1] = *(const v8h*)(pB + 32 * KP + (KO_) + 16);                 \
        (B3_).h[0] = *(const v8h*)(pB + 48 * KP + (KO_));                      \
        (B3_).h[1] = *(const v8h*)(pB + 48 * KP + (KO_) + 16);                 \
    } while (0)

__global__ void k_gemm1(const _Float16* __restrict__ Xh,
                        const _Float16* __restrict__ Wt,
                        float* __restrict__ Ih) {
    const int wave  = (blockIdx.x * blockDim.x + threadIdx.x) >> 5;
    const int lane  = threadIdx.x & 31;
    const int mtile = wave >> 2;
    const int n0    = (wave & 3) << 6;                 // 0,64,128,192
    const int row   = lane & 15;
    const int hi    = lane >> 4;                       // K-half select (ISA layout)

    const _Float16* pA = Xh + (size_t)(mtile * 16 + row) * KP + hi * 8;
    const _Float16* pB = Wt + (size_t)(n0 + row) * KP + hi * 8;

    v8f c0 = {}, c1 = {}, c2 = {}, c3 = {};
    V16 a, b0, b1, b2, b3;
    G1_LOAD(a, b0, b1, b2, b3, 0);
    for (int kt = 0; kt < 24; ++kt) {
        const int kn = (kt + 1) * 32;
        __builtin_prefetch(pA + kn + 128, 0, 1);       // global_prefetch_b8
        V16 an, bn0, bn1, bn2, bn3;
        G1_LOAD(an, bn0, bn1, bn2, bn3, kn);           // next tile in flight
        c0 = __builtin_amdgcn_wmma_f32_16x16x32_f16(false, a.v, false, b0.v, (short)0, c0, false, false);
        c1 = __builtin_amdgcn_wmma_f32_16x16x32_f16(false, a.v, false, b1.v, (short)0, c1, false, false);
        c2 = __builtin_amdgcn_wmma_f32_16x16x32_f16(false, a.v, false, b2.v, (short)0, c2, false, false);
        c3 = __builtin_amdgcn_wmma_f32_16x16x32_f16(false, a.v, false, b3.v, (short)0, c3, false, false);
        a = an; b0 = bn0; b1 = bn1; b2 = bn2; b3 = bn3;
    }
    c0 = __builtin_amdgcn_wmma_f32_16x16x32_f16(false, a.v, false, b0.v, (short)0, c0, false, false);
    c1 = __builtin_amdgcn_wmma_f32_16x16x32_f16(false, a.v, false, b1.v, (short)0, c1, false, false);
    c2 = __builtin_amdgcn_wmma_f32_16x16x32_f16(false, a.v, false, b2.v, (short)0, c2, false, false);
    c3 = __builtin_amdgcn_wmma_f32_16x16x32_f16(false, a.v, false, b3.v, (short)0, c3, false, false);

    // C layout: VGPR r -> M = hi*8 + r ; N = n0 + j*16 + row
    float* pC = Ih + (size_t)(mtile * 16 + hi * 8) * NH + n0 + row;
#pragma unroll
    for (int r = 0; r < 8; ++r) {
        pC[(size_t)r * NH +  0] = c0[r];
        pC[(size_t)r * NH + 16] = c1[r];
        pC[(size_t)r * NH + 32] = c2[r];
        pC[(size_t)r * NH + 48] = c3[r];
    }
}

// ---- hidden LIF recurrence: one thread per (b,h), serial over t ------------
__global__ void k_hidden(const float* __restrict__ Ih, _Float16* __restrict__ Sh) {
    const int idx = blockIdx.x * blockDim.x + threadIdx.x;  // 32768 threads
    const int h = idx & (NH - 1);
    const int b = idx >> 8;
    float v = 0.0f;
    for (int t = 0; t < TT; ++t) {
        size_t m = (size_t)t * BATCH + b;
        float x = Ih[m * NH + h];                       // coalesced over h
        v = 0.9f * v + 0.1f * x;                        // v += (-(v)+I)*dt/tau
        Sh[m * NH + h] = (_Float16)sigm5(v);
    }
}

// ---- GEMM2: I_o(64000x16) = s_h(64000x256) @ Wot^T ; wave = 16x16 tile -----
__global__ void k_gemm2(const _Float16* __restrict__ Sh,
                        const _Float16* __restrict__ Wot,
                        float* __restrict__ Io) {
    const int wave = (blockIdx.x * blockDim.x + threadIdx.x) >> 5;  // = mtile
    const int lane = threadIdx.x & 31;
    const int row  = lane & 15;
    const int hi   = lane >> 4;
    const _Float16* pA = Sh + (size_t)(wave * 16 + row) * NH + hi * 8;
    const _Float16* pB = Wot + (size_t)row * NH + hi * 8;
    v8f c = {};
#pragma unroll
    for (int kt = 0; kt < 8; ++kt) {
        const int ko = kt * 32;
        V16 a, b;
        a.h[0] = *(const v8h*)(pA + ko);
        a.h[1] = *(const v8h*)(pA + ko + 16);
        b.h[0] = *(const v8h*)(pB + ko);
        b.h[1] = *(const v8h*)(pB + ko + 16);
        c = __builtin_amdgcn_wmma_f32_16x16x32_f16(false, a.v, false, b.v, (short)0, c, false, false);
    }
    float* pC = Io + (size_t)(wave * 16 + hi * 8) * 16 + row;
#pragma unroll
    for (int r = 0; r < 8; ++r) pC[(size_t)r * 16] = c[r];
}

// ---- output LIF recurrence + spikes + firing rates -------------------------
__global__ void k_output(const float* __restrict__ Io, float* __restrict__ out) {
    const int idx = blockIdx.x * blockDim.x + threadIdx.x;  // 1280 threads
    const int b = idx / NOUT;
    const int o = idx - b * NOUT;
    float v = 0.0f, sum = 0.0f;
    for (int t = 0; t < TT; ++t) {
        float x = Io[((size_t)t * BATCH + b) * 16 + o];
        v = 0.9f * v + 0.1f * x;
        float s = sigm5(v);
        out[(size_t)b * (NOUT * TT) + (size_t)o * TT + t] = s;
        sum += s;
    }
    out[(size_t)BATCH * NOUT * TT + (size_t)b * NOUT + o] = sum * (1.0f / (float)TT);
}

// ---------------------------------------------------------------------------
// workspace layout (bytes), all 256-aligned; Sh reuses Xh (Xh dead after GEMM1)
//   Wt : 256*800*2    =    409,600  @ 0
//   Wot: 16*256*2     =      8,192  @ 409,600
//   Io : 64000*16*4   =  4,096,000  @ 417,792
//   Ih : 64000*256*4  = 65,536,000  @ 4,513,792
//   Xh : 64000*800*2  =102,400,000  @ 70,049,792   (Sh overlays this region)
// total ~172.5 MB
// ---------------------------------------------------------------------------
extern "C" void kernel_launch(void* const* d_in, const int* in_sizes, int n_in,
                              void* d_out, int out_size, void* d_ws, size_t ws_size,
                              hipStream_t stream) {
    (void)in_sizes; (void)n_in; (void)out_size; (void)ws_size;
    const float* spikes = (const float*)d_in[0];   // (128, 784, 500) f32
    const float* w1     = (const float*)d_in[1];   // (784, 256) f32
    const float* w2     = (const float*)d_in[2];   // (256, 10)  f32
    float* out = (float*)d_out;                    // 640000 spikes + 1280 rates
    char* ws = (char*)d_ws;

    _Float16* Wt  = (_Float16*)(ws + 0);
    _Float16* Wot = (_Float16*)(ws + 409600);
    float*    Io  = (float*)   (ws + 417792);
    float*    Ih  = (float*)   (ws + 4513792);
    _Float16* Xh  = (_Float16*)(ws + 70049792);
    _Float16* Sh  = (_Float16*)(ws + 70049792);    // reuse: Xh consumed by GEMM1

    k_prep_w1 <<<KP,  256, 0, stream>>>(w1, Wt);                   // 800*256
    k_prep_w2 <<<16,  256, 0, stream>>>(w2, Wot);                  // 16*256
    k_transpose<<<dim3(25, 16, BATCH), dim3(32, 8), 0, stream>>>(spikes, Xh);
    k_gemm1   <<<2000, 256, 0, stream>>>(Xh, Wt, Ih);              // 16000 waves
    k_hidden  <<<128, 256, 0, stream>>>(Ih, Sh);                   // 32768 threads
    k_gemm2   <<<500, 256, 0, stream>>>(Sh, Wot, Io);              // 4000 waves
    k_output  <<<5,   256, 0, stream>>>(Io, out);                  // 1280 threads
}
